// TinyNEM_51591147159728
// MI455X (gfx1250) — compile-verified, matrix operand order checked by reading, not run on hardware
//
#include <hip/hip_runtime.h>
#include <hip/hip_bf16.h>
#include <math.h>

// ---------------------------------------------------------------------------
// TinyNEM pipeline for gfx1250 (MI455X, wave32, WMMA + TDM)
//   B=8192, H=512, I=1024, N_REGS=8, R_BIT=16, G=16, L=16, N_OPS=16
// Stage 0: bf16 convert z; bf16-transpose W1, WR         (stream, one-time)
// Stage 1: h  = relu(z @ W1 + b1)     (WMMA bf16, TDM tile staging)
// Stage 2: Rb = sigmoid(h @ WR + bR)  (WMMA bf16, TDM tile staging)
// Stage 2b: threshold = sigmoid(h @ Wt + bt)
// Stage 3: soft register-machine interpreter (VALU)
// Stage 4: LN(R @ W_r2h + b)  -> 268 MB stream (wave-per-row, NT stores)
// ---------------------------------------------------------------------------

#define B_DIM 8192
#define H_DIM 512
#define I_DIM 1024
#define NREG 8
#define RBIT 16
#define GNUM 16
#define LPROG 16

typedef __attribute__((ext_vector_type(16))) __bf16 v16bf;
typedef __attribute__((ext_vector_type(8)))  __bf16 v8bf;
typedef __attribute__((ext_vector_type(8)))  float  v8f;
typedef __attribute__((ext_vector_type(4)))  float  f32x4;
typedef unsigned int u32;
typedef __attribute__((ext_vector_type(4))) u32 u32x4;
typedef __attribute__((ext_vector_type(8))) int  i32x8;
typedef __attribute__((ext_vector_type(4))) int  i32x4;

__device__ __forceinline__ float sigf(float x) {
  return 1.0f / (1.0f + __expf(-x));
}

// ---------------------------------------------------------------------------
// Tensor Data Mover: 2-D bf16 tile (tile_d1 rows x tile_d0 elems, K-major)
// global -> LDS.  D# per CDNA5 ISA 8.3/8.4:
//   group0: count=1 | lds_addr | global_addr | type=2
//   group1: data_size=1 (2B), tensor dims, tile dims, dim0 stride
// 6-arg builtin form (this toolchain): (g0, g1, g2, g3, g_extra, cpol)
// ---------------------------------------------------------------------------
__device__ __forceinline__ void tdm_load_2d_bf16(
    u32 lds_byte_off, const __bf16* gptr,
    u32 tensor_d0, u32 tensor_d1, u32 stride0_elems,
    u32 tile_d0, u32 tile_d1)
{
  unsigned long long ga = (unsigned long long)(uintptr_t)gptr;
  u32x4 g0;
  g0.x = 1u;                                            // count=1 (valid)
  g0.y = lds_byte_off;                                  // lds_addr
  g0.z = (u32)(ga & 0xffffffffu);                       // global_addr[31:0]
  g0.w = (u32)((ga >> 32) & 0x01ffffffu) | (2u << 30);  // ga[56:32] | type=2
  i32x8 g1;
  g1[0] = (int)(1u << 16);                              // data_size=2B
  g1[1] = (int)(tensor_d0 << 16);                       // tensor_dim0 lo16
  g1[2] = (int)((tensor_d0 >> 16) | (tensor_d1 << 16)); // d0 hi | d1 lo
  g1[3] = (int)((tensor_d1 >> 16) | (tile_d0 << 16));   // d1 hi | tile_dim0
  g1[4] = (int)(tile_d1 & 0xffffu);                     // tile_dim1, dim2=0
  g1[5] = (int)stride0_elems;                           // dim0 stride lo32
  g1[6] = 0;
  g1[7] = 0;
  i32x4 gz4 = {0, 0, 0, 0};
  i32x8 gz8 = {0, 0, 0, 0, 0, 0, 0, 0};
  __builtin_amdgcn_tensor_load_to_lds(g0, g1, gz4, gz4, gz8, 0);
}

// ---------------------------------------------------------------------------
// Stage 0a: f32 -> bf16 elementwise (z -> zb). 4 elems/thread, coalesced.
// ---------------------------------------------------------------------------
__global__ __launch_bounds__(256) void convert_bf16_kernel(
    const float* __restrict__ src, __bf16* __restrict__ dst, int n4)
{
  int i = blockIdx.x * 256 + threadIdx.x;
  if (i >= n4) return;
  f32x4 v = *(const f32x4*)(src + (size_t)i * 4);
  __bf16* d = dst + (size_t)i * 4;
  d[0] = (__bf16)v.x; d[1] = (__bf16)v.y;
  d[2] = (__bf16)v.z; d[3] = (__bf16)v.w;
}

// ---------------------------------------------------------------------------
// Stage 0b: f32 [R][C] -> bf16 transpose [C][R] via 32x33 LDS tile.
// ---------------------------------------------------------------------------
__global__ __launch_bounds__(256) void transpose_bf16_kernel(
    const float* __restrict__ src, __bf16* __restrict__ dst, int R, int C)
{
  __shared__ __bf16 t[32][33];
  const int bx = blockIdx.x * 32;     // column block (of src)
  const int by = blockIdx.y * 32;     // row block (of src)
  const int tx = threadIdx.x & 31;
  const int ty = threadIdx.x >> 5;    // 8 rows per pass
  #pragma unroll
  for (int i = 0; i < 32; i += 8)
    t[ty + i][tx] = (__bf16)src[(size_t)(by + ty + i) * C + bx + tx];
  __syncthreads();
  #pragma unroll
  for (int i = 0; i < 32; i += 8)
    dst[(size_t)(bx + ty + i) * R + by + tx] = t[tx][ty + i];
}

// ---------------------------------------------------------------------------
// WMMA GEMM core: C[m0:m0+128, n0:n0+128] over K, A [M][K] bf16 K-major,
// BT [N][K] bf16 K-major (pre-transposed weights). 8 waves, each wave owns
// one 16-row strip x eight 16-col tiles. Tiles staged into LDS by the TDM
// (wave 0 issues descriptors, waits TENSORcnt, block barrier releases).
// Fragments are two contiguous ds_load_b128 each (ISA 7.12.2 16-bit layout:
// per lane-half hf, elems 0..7 <- K=hf*8.., elems 8..15 <- K=16+hf*8..).
// ---------------------------------------------------------------------------
template<int KTOT>
__device__ __forceinline__ void gemm_core(
    const __bf16* __restrict__ A, const __bf16* __restrict__ BT,
    __bf16* lA, __bf16* lB, u32 offA, u32 offB,
    int m0, int n0, int wave, int lane, v8f acc[8])
{
  const int m  = lane & 15;
  const int hf = lane >> 4;
  for (int k0 = 0; k0 < KTOT; k0 += 32) {
    __syncthreads();                      // previous tiles fully consumed
    if (wave == 0) {
      tdm_load_2d_bf16(offA, A + (size_t)m0 * KTOT + k0,
                       (u32)KTOT, 128u, (u32)KTOT, 32u, 128u);
      tdm_load_2d_bf16(offB, BT + (size_t)n0 * KTOT + k0,
                       (u32)KTOT, 128u, (u32)KTOT, 32u, 128u);
      if (k0 + 32 < KTOT)
        __builtin_prefetch(A + (size_t)m0 * KTOT + k0 + 32, 0, 0);
      __builtin_amdgcn_s_wait_tensorcnt(0);
    }
    __syncthreads();                      // tiles visible to all waves

    v8bf alo = *(const v8bf*)&lA[(wave * 16 + m) * 32 + hf * 8];
    v8bf ahi = *(const v8bf*)&lA[(wave * 16 + m) * 32 + 16 + hf * 8];
    v16bf afrag = __builtin_shufflevector(alo, ahi,
        0, 1, 2, 3, 4, 5, 6, 7, 8, 9, 10, 11, 12, 13, 14, 15);
    #pragma unroll
    for (int nt = 0; nt < 8; ++nt) {
      v8bf blo = *(const v8bf*)&lB[(nt * 16 + m) * 32 + hf * 8];
      v8bf bhi = *(const v8bf*)&lB[(nt * 16 + m) * 32 + 16 + hf * 8];
      v16bf bfrag = __builtin_shufflevector(blo, bhi,
          0, 1, 2, 3, 4, 5, 6, 7, 8, 9, 10, 11, 12, 13, 14, 15);
      acc[nt] = __builtin_amdgcn_wmma_f32_16x16x32_bf16(
          false, afrag, false, bfrag, (short)0, acc[nt], false, false);
    }
  }
}

// ---------------------------------------------------------------------------
// Stage 1: h = relu(zb @ W1 + b1) -> bf16
// ---------------------------------------------------------------------------
__global__ __launch_bounds__(256) void gemm1_relu_kernel(
    const __bf16* __restrict__ Zb, const __bf16* __restrict__ W1T,
    const float* __restrict__ b1, __bf16* __restrict__ Hout)
{
  __shared__ __bf16 lA[128 * 32];
  __shared__ __bf16 lB[128 * 32];
  const int lane = threadIdx.x & 31;
  const int wave = threadIdx.x >> 5;
  const int m0 = blockIdx.y * 128;
  const int n0 = blockIdx.x * 128;
  const u32 offA = (u32)(uintptr_t)(void*)lA;   // generic LDS ptr low 32 = offset
  const u32 offB = (u32)(uintptr_t)(void*)lB;

  v8f acc[8] = {};
  gemm_core<H_DIM>(Zb, W1T, lA, lB, offA, offB, m0, n0, wave, lane, acc);

  const int ncol = lane & 15;
  const int hf   = lane >> 4;
  #pragma unroll
  for (int nt = 0; nt < 8; ++nt) {
    #pragma unroll
    for (int r = 0; r < 8; ++r) {
      int mm = m0 + wave * 16 + hf * 8 + r;
      int nn = n0 + nt * 16 + ncol;
      float val = acc[nt][r] + b1[nn];
      val = val > 0.0f ? val : 0.0f;
      Hout[(size_t)mm * I_DIM + nn] = (__bf16)val;
    }
  }
}

// ---------------------------------------------------------------------------
// Stage 2: R_bits = sigmoid(h @ WR + bR) -> f32 [B,128]
// ---------------------------------------------------------------------------
__global__ __launch_bounds__(256) void gemm2_sig_kernel(
    const __bf16* __restrict__ Hbf, const __bf16* __restrict__ WRT,
    const float* __restrict__ bR, float* __restrict__ Rbits)
{
  __shared__ __bf16 lA[128 * 32];
  __shared__ __bf16 lB[128 * 32];
  const int lane = threadIdx.x & 31;
  const int wave = threadIdx.x >> 5;
  const int m0 = blockIdx.y * 128;
  const u32 offA = (u32)(uintptr_t)(void*)lA;
  const u32 offB = (u32)(uintptr_t)(void*)lB;

  v8f acc[8] = {};
  gemm_core<I_DIM>(Hbf, WRT, lA, lB, offA, offB, m0, 0, wave, lane, acc);

  const int ncol = lane & 15;
  const int hf   = lane >> 4;
  #pragma unroll
  for (int nt = 0; nt < 8; ++nt) {
    #pragma unroll
    for (int r = 0; r < 8; ++r) {
      int mm = m0 + wave * 16 + hf * 8 + r;
      int nn = nt * 16 + ncol;
      Rbits[(size_t)mm * 128 + nn] = sigf(acc[nt][r] + bR[nn]);
    }
  }
}

// ---------------------------------------------------------------------------
// Stage 2b: threshold = sigmoid(h @ Wt + bt). One block per batch row.
// ---------------------------------------------------------------------------
__global__ __launch_bounds__(256) void threshold_kernel(
    const __bf16* __restrict__ Hbf, const float* __restrict__ Wt,
    const float* __restrict__ bt, float* __restrict__ Tout)
{
  __shared__ float red[256];
  const int row = blockIdx.x;
  const int tid = threadIdx.x;
  float p = 0.0f;
  #pragma unroll
  for (int j = 0; j < I_DIM / 256; ++j) {
    int c = tid + j * 256;
    p += (float)Hbf[(size_t)row * I_DIM + c] * Wt[c];
  }
  red[tid] = p;
  __syncthreads();
  for (int s = 128; s > 0; s >>= 1) {
    if (tid < s) red[tid] += red[tid + s];
    __syncthreads();
  }
  if (tid == 0) Tout[row] = sigf(red[0] + bt[0]);
}

// ---------------------------------------------------------------------------
// Stage 3: soft register-machine interpreter. One thread per (b, g).
// ---------------------------------------------------------------------------
__device__ __forceinline__ unsigned hash_u32(unsigned x) {
  x ^= x >> 17; x *= 0xed5ad4bbu;
  x ^= x >> 11; x *= 0xac4c1b51u;
  x ^= x >> 15; x *= 0x31848babu;
  x ^= x >> 14;
  return x;
}

__global__ __launch_bounds__(256) void interp_kernel(
    const float* __restrict__ Rbits,
    const float* __restrict__ P_op,  const float* __restrict__ P_dst,
    const float* __restrict__ P_s1,  const float* __restrict__ P_s2,
    const float* __restrict__ P_imm, const float* __restrict__ Proglen,
    float* __restrict__ Rreg)
{
  const int idx = blockIdx.x * 256 + threadIdx.x;   // b*G + g
  const int b = idx >> 4;
  const int g = idx & 15;

  float Rb[NREG][RBIT];
  #pragma unroll
  for (int r = 0; r < NREG; ++r)
    #pragma unroll
    for (int k = 0; k < RBIT; ++k)
      Rb[r][k] = Rbits[(size_t)b * 128 + r * RBIT + k];

  // FLIP=2 stochastic bit flips for groups g>0 (deterministic hash PRNG).
  if (g != 0) {
    unsigned id0 = hash_u32(((unsigned)b * 1103515245u) ^
                            ((unsigned)g * 2654435769u) ^ 0x9E3779B9u) & 127u;
    unsigned id1 = hash_u32(((unsigned)b * 1103515245u) ^
                            ((unsigned)g * 2654435769u) ^ 0x85EBCA6Bu) & 127u;
    Rb[id0 >> 4][id0 & 15] = 1.0f - Rb[id0 >> 4][id0 & 15];
    if (id1 != id0)
      Rb[id1 >> 4][id1 & 15] = 1.0f - Rb[id1 >> 4][id1 & 15];
  }

  float R[NREG], M[NREG];
  #pragma unroll
  for (int r = 0; r < NREG; ++r) {
    float s = 0.0f;
    #pragma unroll
    for (int k = 0; k < RBIT; ++k) s = fmaf(Rb[r][k], (float)(1u << k), s);
    R[r] = s;
    M[r] = 0.0f;
  }

  float plen = 0.0f;
  #pragma unroll
  for (int k = 0; k < 4; ++k)
    plen = fmaf(Proglen[(size_t)b * 4 + k], (float)(1u << k), plen);

  #pragma unroll 1
  for (int t = 0; t < LPROG; ++t) {
    const float* po = &P_op [((size_t)b * LPROG + t) * 4];
    const float* pd = &P_dst[((size_t)b * LPROG + t) * 3];
    const float* p1 = &P_s1 [((size_t)b * LPROG + t) * 3];
    const float* p2 = &P_s2 [((size_t)b * LPROG + t) * 3];
    const float* pi = &P_imm[((size_t)b * LPROG + t) * 3];
    float op  = po[0] + 2.f * po[1] + 4.f * po[2] + 8.f * po[3];
    float dst = pd[0] + 2.f * pd[1] + 4.f * pd[2];
    float s1  = p1[0] + 2.f * p1[1] + 4.f * p1[2];
    float s2  = p2[0] + 2.f * p2[1] + 4.f * p2[2];
    float imm = pi[0] + 2.f * pi[1] + 4.f * pi[2];

    float active = fminf(fmaxf(plen - (float)t, 0.0f), 1.0f);

    float w_d[NREG], w_1[NREG], w_2[NREG];
    float r1 = 0.f, r2 = 0.f, m1 = 0.f, oldv = 0.f;
    #pragma unroll
    for (int i = 0; i < NREG; ++i) {
      w_d[i] = fmaxf(1.0f - fabsf(dst - (float)i), 0.0f);
      w_1[i] = fmaxf(1.0f - fabsf(s1  - (float)i), 0.0f);
      w_2[i] = fmaxf(1.0f - fabsf(s2  - (float)i), 0.0f);
      r1   = fmaf(w_1[i], R[i], r1);
      r2   = fmaf(w_2[i], R[i], r2);
      m1   = fmaf(w_1[i], M[i], m1);
      oldv = fmaf(w_d[i], R[i], oldv);
    }

    float w_op[16];
    #pragma unroll
    for (int o = 0; o < 16; ++o)
      w_op[o] = fmaxf(1.0f - fabsf(op - (float)o), 0.0f);

    float res = 0.f;
    res = fmaf(w_op[0],  oldv, res);
    res = fmaf(w_op[1],  r1 + r2, res);
    res = fmaf(w_op[2],  r1 - r2, res);
    res = fmaf(w_op[3],  r1 * r2 * (1.0f / 65536.0f), res);
    res = fmaf(w_op[4],  r1, res);
    res = fmaf(w_op[5],  imm, res);
    res = fmaf(w_op[6],  r1 + imm, res);
    res = fmaf(w_op[7],  fmaxf(r1, r2), res);
    res = fmaf(w_op[8],  fminf(r1, r2), res);
    res = fmaf(w_op[9],  -r1, res);
    res = fmaf(w_op[10], 0.5f * r1, res);
    res = fmaf(w_op[11], r2, res);
    res = fmaf(w_op[12], m1, res);
    res = fmaf(w_op[13], oldv, res);
    res = fmaf(w_op[14], fabsf(r1), res);
    res = fmaf(w_op[15], oldv, res);

    const float p_store = w_op[13];
    #pragma unroll
    for (int i = 0; i < NREG; ++i) {
      R[i] = fmaf(active * w_d[i], res - R[i], R[i]);
      M[i] = fmaf(active * p_store * w_d[i], r1 - M[i], M[i]);
    }
  }

  #pragma unroll
  for (int r = 0; r < NREG; ++r)
    Rreg[(size_t)idx * NREG + r] = R[r];
}

// ---------------------------------------------------------------------------
// Stage 4: reg_hidden = LayerNorm(R @ W_r2h + b_r2h). One wave32 per row,
// 16 cols per lane, __shfl_xor reductions, nontemporal streaming stores.
// 268 MB written => the HBM roofline floor (~11.5 us @ 23.3 TB/s).
// ---------------------------------------------------------------------------
__global__ __launch_bounds__(256) void r2h_ln_kernel(
    const float* __restrict__ Rreg, const float* __restrict__ W_r2h,
    const float* __restrict__ b_r2h, const float* __restrict__ ln_g,
    const float* __restrict__ ln_b, float* __restrict__ Out)
{
  const int row  = blockIdx.x * 8 + (threadIdx.x >> 5);
  const int lane = threadIdx.x & 31;

  float rr[NREG];
  #pragma unroll
  for (int k = 0; k < NREG; ++k) rr[k] = Rreg[(size_t)row * NREG + k];

  float v[16];
  float sum = 0.0f;
  #pragma unroll
  for (int j = 0; j < 16; ++j) {
    const int c = j * 32 + lane;
    float x = b_r2h[c];
    #pragma unroll
    for (int k = 0; k < NREG; ++k)
      x = fmaf(rr[k], W_r2h[(size_t)k * H_DIM + c], x);
    v[j] = x;
    sum += x;
  }
  #pragma unroll
  for (int s = 16; s > 0; s >>= 1) sum += __shfl_xor(sum, s, 32);
  const float mu = sum * (1.0f / 512.0f);

  float vs = 0.0f;
  #pragma unroll
  for (int j = 0; j < 16; ++j) {
    const float d = v[j] - mu;
    vs = fmaf(d, d, vs);
  }
  #pragma unroll
  for (int s = 16; s > 0; s >>= 1) vs += __shfl_xor(vs, s, 32);
  const float inv = rsqrtf(vs * (1.0f / 512.0f) + 1e-5f);

  #pragma unroll
  for (int j = 0; j < 16; ++j) {
    const int c = j * 32 + lane;
    const float y = (v[j] - mu) * inv * ln_g[c] + ln_b[c];
    __builtin_nontemporal_store(y, &Out[(size_t)row * H_DIM + c]);
  }
}

// ---------------------------------------------------------------------------
// Workspace layout:
//   [0,  8MB)   zb   bf16  8192*512
//   [8, 24MB)   h    bf16  8192*1024
//   [24, 25MB)  W1T  bf16  1024*512
//   [25, 25.25) WRT  bf16  128*1024
//   [25.25, 29.25)  R_bits f32 8192*128
//   [29.25, 33.25)  Rreg f32   131072*8
// ---------------------------------------------------------------------------
extern "C" void kernel_launch(void* const* d_in, const int* in_sizes, int n_in,
                              void* d_out, int out_size, void* d_ws, size_t ws_size,
                              hipStream_t stream) {
  const float* z_hidden = (const float*)d_in[0];
  const float* P_op     = (const float*)d_in[1];
  const float* P_dst    = (const float*)d_in[2];
  const float* P_src1   = (const float*)d_in[3];
  const float* P_src2   = (const float*)d_in[4];
  const float* P_imm    = (const float*)d_in[5];
  const float* Proglen  = (const float*)d_in[6];
  // d_in[7] = group_num (int scalar) == 16, baked into GNUM
  const float* W1    = (const float*)d_in[8];
  const float* b1    = (const float*)d_in[9];
  const float* WR    = (const float*)d_in[10];
  const float* bR    = (const float*)d_in[11];
  const float* Wt    = (const float*)d_in[12];
  const float* bt    = (const float*)d_in[13];
  const float* W_r2h = (const float*)d_in[14];
  const float* b_r2h = (const float*)d_in[15];
  const float* ln_g  = (const float*)d_in[16];
  const float* ln_b  = (const float*)d_in[17];

  char* ws = (char*)d_ws;
  size_t off = 0;
  __bf16* zb_ws  = (__bf16*)(ws + off); off += (size_t)B_DIM * H_DIM * 2;
  __bf16* h_ws   = (__bf16*)(ws + off); off += (size_t)B_DIM * I_DIM * 2;
  __bf16* w1t_ws = (__bf16*)(ws + off); off += (size_t)I_DIM * H_DIM * 2;
  __bf16* wrt_ws = (__bf16*)(ws + off); off += (size_t)128 * I_DIM * 2;
  float*  rb_ws  = (float*)(ws + off);  off += (size_t)B_DIM * 128 * 4;
  float*  rreg_ws= (float*)(ws + off);

  float* out_hidden = (float*)d_out;                              // [B*G, 512]
  float* out_thresh = out_hidden + (size_t)B_DIM * GNUM * H_DIM;  // [B]

  // 0: operand conditioning (bf16 convert + weight transposes)
  convert_bf16_kernel<<<dim3((B_DIM * H_DIM / 4) / 256), 256, 0, stream>>>(
      z_hidden, zb_ws, B_DIM * H_DIM / 4);
  transpose_bf16_kernel<<<dim3(I_DIM / 32, H_DIM / 32), 256, 0, stream>>>(
      W1, w1t_ws, H_DIM, I_DIM);
  transpose_bf16_kernel<<<dim3(128 / 32, I_DIM / 32), 256, 0, stream>>>(
      WR, wrt_ws, I_DIM, 128);

  // 1: h = relu(zb @ W1 + b1)
  gemm1_relu_kernel<<<dim3(I_DIM / 128, B_DIM / 128), 256, 0, stream>>>(
      zb_ws, w1t_ws, b1, h_ws);

  // 2: R_bits = sigmoid(h @ WR + bR)
  gemm2_sig_kernel<<<dim3(1, B_DIM / 128), 256, 0, stream>>>(
      h_ws, wrt_ws, bR, rb_ws);

  // 2b: threshold = sigmoid(h @ Wt + bt)
  threshold_kernel<<<dim3(B_DIM), 256, 0, stream>>>(h_ws, Wt, bt, out_thresh);

  // 3: interpreter over B*G lanes
  interp_kernel<<<dim3((B_DIM * GNUM) / 256), 256, 0, stream>>>(
      rb_ws, P_op, P_dst, P_src1, P_src2, P_imm, Proglen, rreg_ws);

  // 4: reg_hidden = LN(R @ W_r2h + b_r2h)
  r2h_ln_kernel<<<dim3((B_DIM * GNUM) / 8), 256, 0, stream>>>(
      rreg_ws, W_r2h, b_r2h, ln_g, ln_b, out_hidden);
}
